// DeepSeekV2Attention_26087631356406
// MI455X (gfx1250) — matrix operand order repaired and checked by reading, not compiled, not run on hardware
//
#include <hip/hip_runtime.h>
#include <hip/hip_bf16.h>
#include <stdint.h>

typedef __attribute__((ext_vector_type(16))) _Float16 v16h;
typedef __attribute__((ext_vector_type(8)))  _Float16 v8h;
typedef __attribute__((ext_vector_type(4)))  _Float16 v4h;
typedef __attribute__((ext_vector_type(8)))  float    v8f;

#define WMMA_F32_F16(a, b, c) \
    __builtin_amdgcn_wmma_f32_16x16x32_f16(false, (a), false, (b), (short)0, (c), false, false)

static constexpr int Bb   = 2;
static constexpr int Ss   = 2048;
static constexpr int HIDc = 2048;
static constexpr int NHh  = 16;
static constexpr int DNc  = 128;
static constexpr int DRc  = 64;
static constexpr int DVc  = 128;
static constexpr int QDc  = 192;   // DN + DR
static constexpr int QLRc = 1536;
static constexpr int KVLRc = 512;

// ---- CDNA5 async global->LDS copy (ASYNCcnt-tracked, 16B per lane) ----
__device__ __forceinline__ void async_ld_b128(uint32_t lds_off, const _Float16* gaddr) {
    asm volatile("global_load_async_to_lds_b128 %0, %1, off"
                 :: "v"(lds_off), "v"(gaddr) : "memory");
}
__device__ __forceinline__ void wait_async0() {
    asm volatile("s_wait_asynccnt 0x0" ::: "memory");
}

__device__ __forceinline__ v16h pack16(v8h lo, v8h hi) {
    v16h f;
#pragma unroll
    for (int i = 0; i < 8; i++) { f[i] = lo[i]; f[i + 8] = hi[i]; }
    return f;
}

// A-fragment (16x32 f16, MxK): lane&15 = M, halves 0-7 at K=8*ksel, 8-15 at K=8*ksel+16.
__device__ __forceinline__ v16h load_a_frag(const _Float16* base, int stride, int lane) {
    const int m  = lane & 15;
    const int ks = (lane >> 4) << 3;
    const _Float16* p = base + m * stride + ks;
    v8h lo = *(const v8h*)(p);
    v8h hi = *(const v8h*)(p + 16);
    return pack16(lo, hi);
}
// B-fragment (32x16 f16, KxN) stored N-major: lane&15 = N, halves 0-7 at K=8*ksel, 8-15 at K=8*ksel+16.
__device__ __forceinline__ v16h load_b_frag(const _Float16* base, int stride, int lane) {
    const int n  = lane & 15;
    const int ks = (lane >> 4) << 3;
    const _Float16* p = base + n * stride + ks;
    v8h lo = *(const v8h*)(p);
    v8h hi = *(const v8h*)(p + 16);
    return pack16(lo, hi);
}

// ---------------------------------------------------------------------------
// f32 -> f16 conversion (vectorized, n must be divisible by 4)
// ---------------------------------------------------------------------------
__global__ __launch_bounds__(256) void cvt_f32_to_f16(
    const float* __restrict__ x, _Float16* __restrict__ y, long n4)
{
    long i = (long)blockIdx.x * 256 + threadIdx.x;
    if (i < n4) {
        float4 v = ((const float4*)x)[i];
        v4h h; h[0] = (_Float16)v.x; h[1] = (_Float16)v.y;
               h[2] = (_Float16)v.z; h[3] = (_Float16)v.w;
        ((v4h*)y)[i] = h;
    }
}

// ---------------------------------------------------------------------------
// f16 WMMA GEMM: C[M,N] = A[M,K] * Bw[N,K]^T (+bias[N]), f32 out.
// BK=64 halves per stage; double-buffered LDS fed by async global->LDS b128.
// Block = 256 threads (8 waves, 4(M) x 2(N)), wave tile 32x64.
// Requires M%128==0, K%64==0.
// ---------------------------------------------------------------------------
#define LDAs 72   // padded LDS stride in halves (144B, 16B aligned)

__global__ __launch_bounds__(256) void gemm_bias_wmma(
    const _Float16* __restrict__ A, const _Float16* __restrict__ Bw,
    const float* __restrict__ bias, float* __restrict__ C,
    int M, int N, int K)
{
    __shared__ _Float16 As[2][128 * LDAs];
    __shared__ _Float16 Bs[2][128 * LDAs];

    const int tid  = threadIdx.x;
    const int lane = tid & 31;
    const int wid  = tid >> 5;
    const int wm   = wid & 3;
    const int wn   = wid >> 2;
    const int n0   = blockIdx.x * 128;
    const int m0   = blockIdx.y * 128;

    // per-thread staging addresses: 4 chunks of 16B for A, 4 for B
    const _Float16* aptr[4];
    const _Float16* bptr[4];
    uint32_t loff[4];
#pragma unroll
    for (int i = 0; i < 4; i++) {
        int c   = tid + i * 256;      // 1024 chunks = 128 rows x 8x16B
        int row = c >> 3;
        int chk = c & 7;
        aptr[i] = A + (size_t)(m0 + row) * K + chk * 8;
        int gr  = n0 + row; if (gr > N - 1) gr = N - 1;   // clamp (cols >= N discarded)
        bptr[i] = Bw + (size_t)gr * K + chk * 8;
        loff[i] = (uint32_t)(row * LDAs + chk * 8) * 2;
    }
    const uint32_t asB[2] = { (uint32_t)(uintptr_t)&As[0][0], (uint32_t)(uintptr_t)&As[1][0] };
    const uint32_t bsB[2] = { (uint32_t)(uintptr_t)&Bs[0][0], (uint32_t)(uintptr_t)&Bs[1][0] };

#define ISSUE_TILE(ab, bb)                                   \
    do {                                                     \
        _Pragma("unroll")                                    \
        for (int i = 0; i < 4; i++) {                        \
            async_ld_b128((ab) + loff[i], aptr[i]);          \
            async_ld_b128((bb) + loff[i], bptr[i]);          \
            aptr[i] += 64; bptr[i] += 64;                    \
        }                                                    \
    } while (0)

    v8f acc[2][4] = {};
    const int nk = K >> 6;        // K/64 stages

    ISSUE_TILE(asB[0], bsB[0]);
    wait_async0();
    __syncthreads();

    int cur = 0;
    for (int kt = 0; kt < nk; kt++) {
        if (kt + 1 < nk) { ISSUE_TILE(asB[cur ^ 1], bsB[cur ^ 1]); }

        const _Float16* as = cur ? &As[1][0] : &As[0][0];
        const _Float16* bs = cur ? &Bs[1][0] : &Bs[0][0];
#pragma unroll
        for (int kk = 0; kk < 2; kk++) {
            v16h a0 = load_a_frag(as + (wm * 32 + 0)  * LDAs + kk * 32, LDAs, lane);
            v16h a1 = load_a_frag(as + (wm * 32 + 16) * LDAs + kk * 32, LDAs, lane);
#pragma unroll
            for (int j = 0; j < 4; j++) {
                v16h b = load_b_frag(bs + (wn * 64 + j * 16) * LDAs + kk * 32, LDAs, lane);
                acc[0][j] = WMMA_F32_F16(a0, b, acc[0][j]);
                acc[1][j] = WMMA_F32_F16(a1, b, acc[1][j]);
            }
        }
        if (kt + 1 < nk) wait_async0();   // next tile landed
        __syncthreads();                  // all waves done reading cur
        cur ^= 1;
    }
#undef ISSUE_TILE

    const int rsel = (lane >> 4) << 3;
    const int nl   = lane & 15;
#pragma unroll
    for (int i = 0; i < 2; i++) {
#pragma unroll
        for (int j = 0; j < 4; j++) {
            int col = n0 + wn * 64 + j * 16 + nl;
            if (col < N) {
                float bv = bias ? bias[col] : 0.0f;
#pragma unroll
                for (int r = 0; r < 8; r++) {
                    int row = m0 + wm * 32 + i * 16 + r + rsel;
                    C[(size_t)row * N + col] = acc[i][j][r] + bv;
                }
            }
        }
    }
}

// ---------------------------------------------------------------------------
// Row-wise RMSNorm
// ---------------------------------------------------------------------------
__global__ __launch_bounds__(256) void rmsnorm_kernel(
    const float* __restrict__ x, int xs,
    const float* __restrict__ w,
    float* __restrict__ y, int ys, int L)
{
    __shared__ float red[256];
    const int row = blockIdx.x;
    const float* xr = x + (size_t)row * xs;
    float s = 0.f;
    for (int i = threadIdx.x; i < L; i += 256) { float v = xr[i]; s += v * v; }
    red[threadIdx.x] = s;
    __syncthreads();
    for (int o = 128; o > 0; o >>= 1) {
        if (threadIdx.x < o) red[threadIdx.x] += red[threadIdx.x + o];
        __syncthreads();
    }
    const float scale = rsqrtf(red[0] / (float)L + 1e-6f);
    float* yr = y + (size_t)row * ys;
    for (int i = threadIdx.x; i < L; i += 256) yr[i] = xr[i] * scale * w[i];
}

// ---------------------------------------------------------------------------
// RoPE + pack to f16 attention layouts.
//  Qf/Kf: [B*NH, S, 192] (d contiguous)   Vt: [B*NH, 128, S] (s contiguous)
// ---------------------------------------------------------------------------
__global__ __launch_bounds__(256) void rope_pack_kernel(
    const float* __restrict__ q,     // [B*S, NH*192]
    const float* __restrict__ kvb,   // [B*S, NH*256]
    const float* __restrict__ kva,   // [B*S, 576]
    const int*   __restrict__ pos,   // [B*S]
    _Float16* __restrict__ Qf,
    _Float16* __restrict__ Kf,
    _Float16* __restrict__ Vt)
{
    __shared__ float cs[64], sn[64], kp[64];
    const int row  = blockIdx.x;        // b*S + s
    const int b    = row / Ss;
    const int sidx = row % Ss;
    const int tid  = threadIdx.x;

    if (tid < 32) {
        float invf = __powf(10000.0f, -((float)(2 * tid)) / 64.0f);
        float f = (float)pos[row] * invf;
        float c = __cosf(f), s_ = __sinf(f);
        cs[tid] = c; cs[tid + 32] = c;
        sn[tid] = s_; sn[tid + 32] = s_;
    }
    __syncthreads();
    if (tid < 64) {
        int d = tid;
        float x   = kva[(size_t)row * 576 + 512 + d];
        float rot = (d < 32) ? -kva[(size_t)row * 576 + 512 + d + 32]
                             :  kva[(size_t)row * 576 + 512 + d - 32];
        kp[d] = x * cs[d] + rot * sn[d];
    }
    __syncthreads();

    for (int idx = tid; idx < NHh * QDc; idx += 256) {
        int h = idx / QDc, d = idx % QDc;
        const float* qr = q + (size_t)row * (NHh * QDc) + h * QDc;
        float v;
        if (d < DNc) v = qr[d];
        else {
            int dp = d - DNc;
            float x   = qr[DNc + dp];
            float rot = (dp < 32) ? -qr[DNc + dp + 32] : qr[DNc + dp - 32];
            v = x * cs[dp] + rot * sn[dp];
        }
        Qf[((size_t)(b * NHh + h) * Ss + sidx) * QDc + d] = (_Float16)v;
    }
    for (int idx = tid; idx < NHh * QDc; idx += 256) {
        int h = idx / QDc, d = idx % QDc;
        float v = (d < DNc) ? kvb[(size_t)row * (NHh * 256) + h * 256 + d] : kp[d - DNc];
        Kf[((size_t)(b * NHh + h) * Ss + sidx) * QDc + d] = (_Float16)v;
    }
    for (int idx = tid; idx < NHh * DVc; idx += 256) {
        int h = idx / DVc, d = idx % DVc;
        float v = kvb[(size_t)row * (NHh * 256) + h * 256 + DNc + d];
        Vt[((size_t)(b * NHh + h) * DVc + d) * Ss + sidx] = (_Float16)v;
    }
}

// ---------------------------------------------------------------------------
// Flash attention (causal). 8 waves/block, each wave owns 16 query rows.
// 32 keys/step: 12 WMMA scores (two interleaved chains), LDS bounce, 8 WMMA PV.
// ---------------------------------------------------------------------------
__global__ __launch_bounds__(256) void mla_flash_attn(
    const _Float16* __restrict__ Qf,
    const _Float16* __restrict__ Kf,
    const _Float16* __restrict__ Vt,
    float* __restrict__ Out)          // [B*S, NH*128]
{
    __shared__ _Float16 Plds[8][16 * 32];

    const int tid  = threadIdx.x;
    const int lane = tid & 31;
    const int wid  = tid >> 5;
    const int bh   = blockIdx.y;
    const int b    = bh >> 4;
    const int h    = bh & 15;
    const int q0   = blockIdx.x * 128 + wid * 16;

    const _Float16* Qb = Qf + (size_t)bh * Ss * QDc;
    const _Float16* Kb = Kf + (size_t)bh * Ss * QDc;
    const _Float16* Vb = Vt + (size_t)bh * DVc * Ss;

    const int ml   = lane & 15;
    const int hsel = lane >> 4;

    v16h qfr[6];
    {
        const _Float16* qr = Qb + (size_t)(q0 + ml) * QDc + (hsel << 3);
#pragma unroll
        for (int c = 0; c < 6; c++) {
            v8h lo = *(const v8h*)(qr + c * 32);
            v8h hi = *(const v8h*)(qr + c * 32 + 16);
            qfr[c] = pack16(lo, hi);
        }
    }

    v8f oacc[8] = {};
    float mrun[8], lrun[8];
#pragma unroll
    for (int r = 0; r < 8; r++) { mrun[r] = -1e30f; lrun[r] = 0.0f; }
    const float scale = 0.07216878364870323f;   // 1/sqrt(192)

    _Float16* pl = &Plds[wid][0];

    for (int k0 = 0; k0 < q0 + 16; k0 += 32) {
        // ---- scores: two interleaved WMMA chains over 6 k-chunks ----
        v8f sc[2] = {};
        int kn0 = k0 + ml;      int kc0 = kn0 < Ss ? kn0 : Ss - 1;
        int kn1 = k0 + 16 + ml; int kc1 = kn1 < Ss ? kn1 : Ss - 1;
        const _Float16* kr0 = Kb + (size_t)kc0 * QDc + (hsel << 3);
        const _Float16* kr1 = Kb + (size_t)kc1 * QDc + (hsel << 3);
#pragma unroll
        for (int c = 0; c < 6; c++) {
            v8h lo0 = *(const v8h*)(kr0 + c * 32), hi0 = *(const v8h*)(kr0 + c * 32 + 16);
            v8h lo1 = *(const v8h*)(kr1 + c * 32), hi1 = *(const v8h*)(kr1 + c * 32 + 16);
            sc[0] = WMMA_F32_F16(qfr[c], pack16(lo0, hi0), sc[0]);
            sc[1] = WMMA_F32_F16(qfr[c], pack16(lo1, hi1), sc[1]);
        }
        // ---- scale + causal mask ----
#pragma unroll
        for (int t = 0; t < 2; t++) {
#pragma unroll
            for (int r = 0; r < 8; r++) {
                int qr_ = q0 + r + (hsel << 3);
                int kn  = k0 + 16 * t + ml;
                float v = sc[t][r] * scale;
                sc[t][r] = (kn <= qr_) ? v : -1e30f;
            }
        }
        // ---- online softmax ----
#pragma unroll
        for (int r = 0; r < 8; r++) {
            float mt = fmaxf(sc[0][r], sc[1][r]);
#pragma unroll
            for (int off = 1; off < 16; off <<= 1)
                mt = fmaxf(mt, __shfl_xor(mt, off, 16));
            float mnew  = fmaxf(mrun[r], mt);
            float alpha = __expf(mrun[r] - mnew);
            float p0 = __expf(sc[0][r] - mnew);
            float p1 = __expf(sc[1][r] - mnew);
            sc[0][r] = p0; sc[1][r] = p1;
            float rs = p0 + p1;
#pragma unroll
            for (int off = 1; off < 16; off <<= 1)
                rs += __shfl_xor(rs, off, 16);
            lrun[r] = lrun[r] * alpha + rs;
            mrun[r] = mnew;
#pragma unroll
            for (int j = 0; j < 8; j++) oacc[j][r] *= alpha;
        }
        // ---- P: C-layout -> A-fragment via wave-private LDS ----
#pragma unroll
        for (int t = 0; t < 2; t++)
#pragma unroll
            for (int r = 0; r < 8; r++)
                pl[(r + (hsel << 3)) * 32 + 16 * t + ml] = (_Float16)sc[t][r];
        asm volatile("s_wait_dscnt 0x0" ::: "memory");
        v16h pf;
        {
            const _Float16* pr = pl + ml * 32 + (hsel << 3);
            v8h lo = *(const v8h*)(pr);
            v8h hi = *(const v8h*)(pr + 16);
            pf = pack16(lo, hi);
        }
        // ---- O += P @ V ----
        int ks = k0 + (hsel << 3);
        if (ks > Ss - 24) ks = Ss - 24;   // masked columns are exact zeros
#pragma unroll
        for (int j = 0; j < 8; j++) {
            const _Float16* vr = Vb + (size_t)(16 * j + ml) * Ss + ks;
            v8h lo = *(const v8h*)(vr);
            v8h hi = *(const v8h*)(vr + 16);
            oacc[j] = WMMA_F32_F16(pf, pack16(lo, hi), oacc[j]);
        }
    }

    float inv[8];
#pragma unroll
    for (int r = 0; r < 8; r++) inv[r] = 1.0f / lrun[r];
    float* ob = Out + (size_t)b * Ss * (NHh * DVc) + (size_t)h * DVc;
#pragma unroll
    for (int j = 0; j < 8; j++) {
#pragma unroll
        for (int r = 0; r < 8; r++) {
            int sq = q0 + r + (hsel << 3);
            ob[(size_t)sq * (NHh * DVc) + 16 * j + ml] = oacc[j][r] * inv[r];
        }
    }
}

// ---------------------------------------------------------------------------
extern "C" void kernel_launch(void* const* d_in, const int* in_sizes, int n_in,
                              void* d_out, int out_size, void* d_ws, size_t ws_size,
                              hipStream_t stream)
{
    (void)in_sizes; (void)n_in; (void)out_size; (void)ws_size;
    const float* hs       = (const float*)d_in[0];
    const int*   pos      = (const int*)d_in[2];
    const float* W_qa     = (const float*)d_in[3];
    const float* b_qa     = (const float*)d_in[4];
    const float* w_qa_ln  = (const float*)d_in[5];
    const float* W_qb     = (const float*)d_in[6];
    const float* W_kva    = (const float*)d_in[7];
    const float* b_kva    = (const float*)d_in[8];
    const float* w_kva_ln = (const float*)d_in[9];
    const float* W_kvb    = (const float*)d_in[10];
    const float* W_o      = (const float*)d_in[11];

    const int Mrows = Bb * Ss;    // 4096
    char* ws = (char*)d_ws;
    size_t off = 0;
    auto alloc = [&](size_t bytes) -> char* {
        char* p = ws + off;
        off += (bytes + 255) & ~(size_t)255;
        return p;
    };
    // f32 intermediates
    float*    qa    = (float*)alloc((size_t)Mrows * QLRc * 4);
    float*    kva   = (float*)alloc((size_t)Mrows * (KVLRc + DRc) * 4);
    float*    ckv   = (float*)alloc((size_t)Mrows * KVLRc * 4);
    float*    qproj = (float*)alloc((size_t)Mrows * NHh * QDc * 4);
    float*    kvb   = (float*)alloc((size_t)Mrows * NHh * (DNc + DVc) * 4);
    float*    aout  = (float*)alloc((size_t)Mrows * NHh * DVc * 4);
    // f16 attention operands
    _Float16* Qh    = (_Float16*)alloc((size_t)Bb * NHh * Ss * QDc * 2);
    _Float16* Kh    = (_Float16*)alloc((size_t)Bb * NHh * Ss * QDc * 2);
    _Float16* Vh    = (_Float16*)alloc((size_t)Bb * NHh * DVc * Ss * 2);
    // f16 GEMM operands
    _Float16* hs_h   = (_Float16*)alloc((size_t)Mrows * HIDc * 2);
    _Float16* qa_h   = (_Float16*)alloc((size_t)Mrows * QLRc * 2);
    _Float16* ckv_h  = (_Float16*)alloc((size_t)Mrows * KVLRc * 2);
    _Float16* aout_h = (_Float16*)alloc((size_t)Mrows * NHh * DVc * 2);
    _Float16* Wqa_h  = (_Float16*)alloc((size_t)QLRc * HIDc * 2);
    _Float16* Wkva_h = (_Float16*)alloc((size_t)(KVLRc + DRc) * HIDc * 2);
    _Float16* Wqb_h  = (_Float16*)alloc((size_t)NHh * QDc * QLRc * 2);
    _Float16* Wkvb_h = (_Float16*)alloc((size_t)NHh * (DNc + DVc) * KVLRc * 2);
    _Float16* Wo_h   = (_Float16*)alloc((size_t)HIDc * NHh * DVc * 2);
    (void)alloc(4096);   // tail pad

    dim3 blk(256);
    auto cvt = [&](const float* x, _Float16* y, size_t n) {
        long n4 = (long)(n >> 2);
        cvt_f32_to_f16<<<dim3((unsigned)((n4 + 255) / 256)), blk, 0, stream>>>(x, y, n4);
    };

    // ---- convert static operands once per launch ----
    cvt(hs,    hs_h,   (size_t)Mrows * HIDc);
    cvt(W_qa,  Wqa_h,  (size_t)QLRc * HIDc);
    cvt(W_kva, Wkva_h, (size_t)(KVLRc + DRc) * HIDc);
    cvt(W_qb,  Wqb_h,  (size_t)NHh * QDc * QLRc);
    cvt(W_kvb, Wkvb_h, (size_t)NHh * (DNc + DVc) * KVLRc);
    cvt(W_o,   Wo_h,   (size_t)HIDc * NHh * DVc);

    // 1) q_a = hs @ W_qa^T + b_qa
    gemm_bias_wmma<<<dim3((QLRc + 127) / 128, Mrows / 128), blk, 0, stream>>>(
        hs_h, Wqa_h, b_qa, qa, Mrows, QLRc, HIDc);
    // 2) kv_a = hs @ W_kva^T + b_kva
    gemm_bias_wmma<<<dim3((KVLRc + DRc + 127) / 128, Mrows / 128), blk, 0, stream>>>(
        hs_h, Wkva_h, b_kva, kva, Mrows, KVLRc + DRc, HIDc);
    // 3) rmsnorm(q_a) in place, then f16
    rmsnorm_kernel<<<Mrows, blk, 0, stream>>>(qa, QLRc, w_qa_ln, qa, QLRc, QLRc);
    cvt(qa, qa_h, (size_t)Mrows * QLRc);
    // 4) rmsnorm(c_kv) -> ckv (kva kept for k_pe), then f16
    rmsnorm_kernel<<<Mrows, blk, 0, stream>>>(kva, KVLRc + DRc, w_kva_ln, ckv, KVLRc, KVLRc);
    cvt(ckv, ckv_h, (size_t)Mrows * KVLRc);
    // 5) q = q_a @ W_qb^T
    gemm_bias_wmma<<<dim3(NHh * QDc / 128, Mrows / 128), blk, 0, stream>>>(
        qa_h, Wqb_h, nullptr, qproj, Mrows, NHh * QDc, QLRc);
    // 6) kv_b = ckv @ W_kvb^T
    gemm_bias_wmma<<<dim3(NHh * (DNc + DVc) / 128, Mrows / 128), blk, 0, stream>>>(
        ckv_h, Wkvb_h, nullptr, kvb, Mrows, NHh * (DNc + DVc), KVLRc);
    // 7) RoPE + pack f16 Q/K/V^T
    rope_pack_kernel<<<Mrows, blk, 0, stream>>>(qproj, kvb, kva, pos, Qh, Kh, Vh);
    // 8) causal flash attention
    mla_flash_attn<<<dim3(Ss / 128, Bb * NHh), blk, 0, stream>>>(Qh, Kh, Vh, aout);
    // 9) out = attn_out @ W_o^T
    cvt(aout, aout_h, (size_t)Mrows * NHh * DVc);
    gemm_bias_wmma<<<dim3(HIDc / 128, Mrows / 128), blk, 0, stream>>>(
        aout_h, Wo_h, nullptr, (float*)d_out, Mrows, HIDc, NHh * DVc);
}
